// Tensor_PFF_30365418783345
// MI455X (gfx1250) — matrix operand order repaired and checked by reading, not compiled, not run on hardware
//
#include <hip/hip_runtime.h>
#include <math.h>

typedef __attribute__((ext_vector_type(16))) _Float16 v16h;
typedef __attribute__((ext_vector_type(8)))  float    v8f;
typedef __attribute__((ext_vector_type(4)))  int      v4i;

#define WMMA_F16(A, B, C) \
    __builtin_amdgcn_wmma_f32_16x16x32_f16(false, (A), false, (B), (short)0, (C), false, false)

// --- CDNA5 async global->LDS staging (ASYNCcnt path), with sync fallback ----
#if defined(__HIP_DEVICE_COMPILE__) && __has_builtin(__builtin_amdgcn_global_load_async_to_lds_b128)
#define HAVE_ASYNC 1
typedef __attribute__((address_space(1))) v4i* gv4i_p;   // global v4i*
typedef __attribute__((address_space(3))) v4i* lv4i_p;   // LDS v4i*
#else
#define HAVE_ASYNC 0
#endif

#if HAVE_ASYNC
#define ASYNC_WAIT_32() asm volatile("s_wait_asynccnt 0x20" ::: "memory")
#define ASYNC_WAIT_0()  asm volatile("s_wait_asynccnt 0x0" ::: "memory")
#else
#define ASYNC_WAIT_32()
#define ASYNC_WAIT_0()
#endif

// Stage one 32KB weight panel (8192 halves hi + 8192 halves lo) into LDS.
// 64 threads x 16 iters x 2 (hi,lo) x 16B = 32KB; 32 async-load instrs per wave.
__device__ __forceinline__ void stage_panel(_Float16* __restrict__ dst,
                                            const _Float16* __restrict__ hsrc,
                                            const _Float16* __restrict__ lsrc,
                                            int tid)
{
#if HAVE_ASYNC
#pragma unroll
    for (int j = 0; j < 16; ++j) {
        const int i = (tid + j * 64) * 8;
        __builtin_amdgcn_global_load_async_to_lds_b128(
            (gv4i_p)(hsrc + i), (lv4i_p)(dst + i), 0, 0);
        __builtin_amdgcn_global_load_async_to_lds_b128(
            (gv4i_p)(lsrc + i), (lv4i_p)(dst + 8192 + i), 0, 0);
    }
#else
#pragma unroll
    for (int j = 0; j < 16; ++j) {
        const int i = (tid + j * 64) * 8;
        *(uint4*)(dst + i)        = *(const uint4*)(hsrc + i);
        *(uint4*)(dst + 8192 + i) = *(const uint4*)(lsrc + i);
    }
#endif
}

// ---------------------------------------------------------------------------
// Kernel 1: build dense W[j][i] (512x512) from TT cores, one block per layer.
//   W[j,i] = sum_{r1,r2,r3} G1[m1,n1,r1] G2[r1,m2,n2,r2] G3[r2,m3,n3,r3] G4[r3,m4,n4]
//   i = ((m1*4+m2)*8+m3)*4+m4 ,  j = ((n1*4+n2)*8+n3)*4+n4
// ---------------------------------------------------------------------------
__global__ __launch_bounds__(256) void build_w_kernel(
    const float* __restrict__ g1a, const float* __restrict__ g2a,
    const float* __restrict__ g3a, const float* __restrict__ g4a,
    const float* __restrict__ g1b, const float* __restrict__ g2b,
    const float* __restrict__ g3b, const float* __restrict__ g4b,
    float* __restrict__ wf)
{
    __shared__ float sT12[256 * 20];   // 20 KB
    __shared__ float sT34[20 * 512];   // 40 KB
    const int L = blockIdx.x;
    const float* G1 = L ? g1b : g1a;
    const float* G2 = L ? g2b : g2a;
    const float* G3 = L ? g3b : g3a;
    const float* G4 = L ? g4b : g4a;
    const int tid = threadIdx.x;

    for (int i = tid; i < 256 * 20; i += 256) {
        int a = i / 20, r2 = i % 20;
        int n2 = a & 3, m2 = (a >> 2) & 3, n1 = (a >> 4) & 3, m1 = (a >> 6) & 3;
        float s = 0.f;
        for (int r1 = 0; r1 < 20; ++r1)
            s += G1[(m1 * 4 + n1) * 20 + r1] * G2[((r1 * 4 + m2) * 4 + n2) * 20 + r2];
        sT12[i] = s;
    }
    __syncthreads();

    for (int hs = 0; hs < 2; ++hs) {   // m3 = hs*4 + m3l
        for (int i = tid; i < 20 * 512; i += 256) {
            int r2 = i >> 9, bb = i & 511;
            int n4 = bb & 3, m4 = (bb >> 2) & 3, n3 = (bb >> 4) & 7;
            int m3 = hs * 4 + ((bb >> 7) & 3);
            float s = 0.f;
            for (int r3 = 0; r3 < 20; ++r3)
                s += G3[((r2 * 8 + m3) * 8 + n3) * 20 + r3] * G4[(r3 * 4 + m4) * 4 + n4];
            sT34[i] = s;
        }
        __syncthreads();
        for (int i = tid; i < 256 * 512; i += 256) {
            int a = i >> 9, bb = i & 511;
            float s = 0.f;
            for (int r2 = 0; r2 < 20; ++r2)
                s += sT12[a * 20 + r2] * sT34[r2 * 512 + bb];
            int n2 = a & 3, m2 = (a >> 2) & 3, n1 = (a >> 4) & 3, m1 = (a >> 6) & 3;
            int n4 = bb & 3, m4 = (bb >> 2) & 3, n3 = (bb >> 4) & 7;
            int m3 = hs * 4 + ((bb >> 7) & 3);
            int ii = ((m1 * 4 + m2) * 8 + m3) * 4 + m4;
            int jj = ((n1 * 4 + n2) * 8 + n3) * 4 + n4;
            wf[L * 262144 + jj * 512 + ii] = s;
        }
        __syncthreads();
    }
}

// ---------------------------------------------------------------------------
// Kernel 2: split W into hi/lo f16 and pre-swizzle into WMMA B-operand layout.
// B layout (wave32): lane = column (lane&15), halves 0..15 hold
//   K = 32c + (h&7) + (h&8 ? 16 : 0) + (lane<16 ? 0 : 8).
// Layer 0 stored t-major  : slot = ((t*16 + c)*32 + lane)*16 + h   (stage = t)
// Layer 1 stored c-major  : slot = ((c*32 + t)*32 + lane)*16 + h   (stage = 2c+t/16)
// so every 8192-half stage is a contiguous 16KB block for async staging.
// ---------------------------------------------------------------------------
__global__ __launch_bounds__(256) void pack_w_kernel(
    const float* __restrict__ wf,
    _Float16* __restrict__ bhi, _Float16* __restrict__ blo)
{
    unsigned g = blockIdx.x * 256u + threadIdx.x;      // < 524288
    int h    = g & 15;
    int lane = (g >> 4) & 31;
    int c    = (g >> 9) & 15;
    int t    = (g >> 13) & 31;
    int L    = g >> 18;
    int k = 32 * c + (h & 7) + ((h & 8) ? 16 : 0) + ((lane & 16) ? 8 : 0);
    int j = 16 * t + (lane & 15);
    float w = wf[L * 262144 + j * 512 + k];
    _Float16 hi = (_Float16)w;
    _Float16 lo = (_Float16)(w - (float)hi);
    unsigned dst = L ? (262144u + (((unsigned)(c * 32 + t) * 32 + lane) * 16 + h))
                     : g;
    bhi[dst] = hi;
    blo[dst] = lo;
}

// ---------------------------------------------------------------------------
// Kernel 3: fused  out = LN(x + GELU(x W1^T + b1) W2^T + b2)
// 64 threads = 2 waves; each wave owns 16 rows. Weights double-buffered in LDS
// via async global->LDS; h stays in LDS (hi|lo packed); LN in registers.
// LDS: hbuf 64KB + 2x32KB panels = 128KB (of 320KB/WGP).
// ---------------------------------------------------------------------------
__global__ __launch_bounds__(64, 1) void fused_ffn_kernel(
    const float* __restrict__ x,
    const _Float16* __restrict__ bhi0, const _Float16* __restrict__ blo0,
    const _Float16* __restrict__ bhi1, const _Float16* __restrict__ blo1,
    const float* __restrict__ b1, const float* __restrict__ b2,
    const float* __restrict__ gamma, const float* __restrict__ beta,
    float* __restrict__ out)
{
    __shared__ unsigned int hbuf[2][16 * 512];     // 64 KB: h packed (hi|lo<<16)
    __shared__ _Float16 bpan[2][2 * 8192];         // 64 KB: double-buffered panels
    const int tid  = threadIdx.x;
    const int wave = tid >> 5;
    const int lane = tid & 31;
    const int hw   = lane >> 4;          // half-wave: 0 = lanes 0-15
    const int ln   = lane & 15;
    const int g0   = hw ? 8 : 0;         // A-layout K-group offset
    const int rowbase = blockIdx.x * 32 + wave * 16;

    // Kick off stage 0 of layer-1 weights while we convert x.
    stage_panel(bpan[0], bhi0, blo0, tid);

    // ---- Load x panel (16 rows x 512) into A-operand registers, hi/lo f16 ----
    // A layout (16-bit, 16x32): halves 0..7 <- K0+g0+e ; halves 8..15 <- K0+16+g0+e
    v16h axh[16], axl[16];
    {
        const float* xr = x + (size_t)(rowbase + ln) * 512;
        #pragma unroll
        for (int c = 0; c < 16; ++c) {
            const float* p = xr + 32 * c + g0;
            #pragma unroll
            for (int e = 0; e < 8; ++e) {
                float v0 = p[e];
                float v1 = p[16 + e];
                _Float16 h0 = (_Float16)v0;
                _Float16 h1 = (_Float16)v1;
                axh[c][e]     = h0;  axl[c][e]     = (_Float16)(v0 - (float)h0);
                axh[c][8 + e] = h1;  axl[c][8 + e] = (_Float16)(v1 - (float)h1);
            }
        }
    }

    unsigned int* hb = hbuf[wave];

    // ---- Layer 1: h = GELU(x W1^T + b1); stage = output tile t ----
    for (int t = 0; t < 32; ++t) {
        const int ns = t + 1;
        if (ns < 32) {
            stage_panel(bpan[ns & 1], bhi0 + ns * 8192, blo0 + ns * 8192, tid);
            ASYNC_WAIT_32();
        } else {
            ASYNC_WAIT_0();
        }
        __syncthreads();
        const _Float16* P = bpan[t & 1];

        v8f acc = {};
        #pragma unroll
        for (int c = 0; c < 16; ++c) {
            v16h bh = *(const v16h*)(P + (c * 32 + lane) * 16);
            v16h bl = *(const v16h*)(P + 8192 + (c * 32 + lane) * 16);
            acc = WMMA_F16(axh[c], bh, acc);
            acc = WMMA_F16(axh[c], bl, acc);
            acc = WMMA_F16(axl[c], bh, acc);
        }
        float bias = b1[t * 16 + ln];
        #pragma unroll
        for (int r = 0; r < 8; ++r) {
            float v = acc[r] + bias;
            v = 0.5f * v * (1.0f + erff(v * 0.70710678118654752440f)); // exact GELU
            _Float16 vh = (_Float16)v;
            _Float16 vl = (_Float16)(v - (float)vh);
            unsigned u = (unsigned)__builtin_bit_cast(unsigned short, vh)
                       | ((unsigned)__builtin_bit_cast(unsigned short, vl) << 16);
            hb[(r + g0) * 512 + t * 16 + ln] = u;   // h[row][col] packed
        }
        __syncthreads();   // both waves done reading this panel buffer
    }

    // ---- Layer 2: out = h W2^T ; 32 live accumulators, stage = (c, t-half) ----
    v8f acc2[32];
    #pragma unroll
    for (int t = 0; t < 32; ++t) { v8f z = {}; acc2[t] = z; }

    stage_panel(bpan[0], bhi1, blo1, tid);   // stage 0 = (c=0, th=0)

    for (int c = 0; c < 16; ++c) {
        // A chunk from LDS h (row = ln), unpack hi/lo; reused for both t-halves
        v16h ahh, ahl;
        const unsigned int* q = hb + ln * 512 + 32 * c + g0;
        #pragma unroll
        for (int e = 0; e < 8; ++e) {
            unsigned u0 = q[e];
            unsigned u1 = q[16 + e];
            ahh[e]     = __builtin_bit_cast(_Float16, (unsigned short)(u0 & 0xffffu));
            ahl[e]     = __builtin_bit_cast(_Float16, (unsigned short)(u0 >> 16));
            ahh[8 + e] = __builtin_bit_cast(_Float16, (unsigned short)(u1 & 0xffffu));
            ahl[8 + e] = __builtin_bit_cast(_Float16, (unsigned short)(u1 >> 16));
        }
        #pragma unroll
        for (int th = 0; th < 2; ++th) {
            const int s  = 2 * c + th;
            const int ns = s + 1;
            if (ns < 32) {
                stage_panel(bpan[ns & 1], bhi1 + ns * 8192, blo1 + ns * 8192, tid);
                ASYNC_WAIT_32();
            } else {
                ASYNC_WAIT_0();
            }
            __syncthreads();
            const _Float16* P = bpan[s & 1];
            #pragma unroll
            for (int b = 0; b < 16; ++b) {
                v16h bh = *(const v16h*)(P + (b * 32 + lane) * 16);
                v16h bl = *(const v16h*)(P + 8192 + (b * 32 + lane) * 16);
                acc2[th * 16 + b] = WMMA_F16(ahh, bh, acc2[th * 16 + b]);
                acc2[th * 16 + b] = WMMA_F16(ahh, bl, acc2[th * 16 + b]);
                acc2[th * 16 + b] = WMMA_F16(ahl, bh, acc2[th * 16 + b]);
            }
            __syncthreads();
        }
    }

    // ---- bias2 + residual folded into accumulators ----
    const int orow0 = rowbase + g0;   // C/D layout: VGPR r <-> row r + 8*hw
    #pragma unroll
    for (int t = 0; t < 32; ++t) {
        float bias = b2[t * 16 + ln];
        #pragma unroll
        for (int r = 0; r < 8; ++r) {
            float xres = x[(size_t)(orow0 + r) * 512 + t * 16 + ln];
            acc2[t][r] += bias + xres;
        }
    }

    // ---- LayerNorm entirely in registers (half-wave shuffles) ----
    float s[8], sq[8];
    #pragma unroll
    for (int r = 0; r < 8; ++r) { s[r] = 0.f; sq[r] = 0.f; }
    #pragma unroll
    for (int t = 0; t < 32; ++t)
        #pragma unroll
        for (int r = 0; r < 8; ++r) {
            float v = acc2[t][r];
            s[r] += v; sq[r] += v * v;
        }
    #pragma unroll
    for (int r = 0; r < 8; ++r) {
        for (int m = 1; m < 16; m <<= 1) {
            s[r]  += __shfl_xor(s[r],  m, 16);
            sq[r] += __shfl_xor(sq[r], m, 16);
        }
    }
    float mu[8], rstd[8];
    #pragma unroll
    for (int r = 0; r < 8; ++r) {
        mu[r] = s[r] * (1.0f / 512.0f);
        float var = sq[r] * (1.0f / 512.0f) - mu[r] * mu[r];
        rstd[r] = rsqrtf(var + 1e-12f);
    }

    // ---- scale/shift + store ----
    #pragma unroll
    for (int t = 0; t < 32; ++t) {
        int col = t * 16 + ln;
        float ga = gamma[col];
        float be = beta[col];
        #pragma unroll
        for (int r = 0; r < 8; ++r) {
            float v = (acc2[t][r] - mu[r]) * rstd[r] * ga + be;
            out[(size_t)(orow0 + r) * 512 + col] = v;
        }
    }
}

// ---------------------------------------------------------------------------
extern "C" void kernel_launch(void* const* d_in, const int* in_sizes, int n_in,
                              void* d_out, int out_size, void* d_ws, size_t ws_size,
                              hipStream_t stream) {
    const float* x    = (const float*)d_in[0];
    const float* f1g1 = (const float*)d_in[1];
    const float* f1g2 = (const float*)d_in[2];
    const float* f1g3 = (const float*)d_in[3];
    const float* f1g4 = (const float*)d_in[4];
    const float* f1b  = (const float*)d_in[5];
    const float* f2g1 = (const float*)d_in[6];
    const float* f2g2 = (const float*)d_in[7];
    const float* f2g3 = (const float*)d_in[8];
    const float* f2g4 = (const float*)d_in[9];
    const float* f2b  = (const float*)d_in[10];
    const float* gam  = (const float*)d_in[11];
    const float* bet  = (const float*)d_in[12];
    float* out = (float*)d_out;

    // workspace layout: [0,2MB) W f32 (2 layers); [2MB,3MB) Bhi; [3MB,4MB) Blo
    float*    wf  = (float*)d_ws;
    _Float16* bhi = (_Float16*)(wf + 2 * 262144);
    _Float16* blo = bhi + 2 * 262144;

    build_w_kernel<<<2, 256, 0, stream>>>(f1g1, f1g2, f1g3, f1g4,
                                          f2g1, f2g2, f2g3, f2g4, wf);
    pack_w_kernel<<<2048, 256, 0, stream>>>(wf, bhi, blo);

    const int T = in_sizes[0] / 512;            // 65536 rows
    fused_ffn_kernel<<<T / 32, 64, 0, stream>>>(
        x, bhi, blo, bhi + 262144, blo + 262144,
        f1b, f2b, gam, bet, out);
}
